// Model_SGN_21474836480504
// MI455X (gfx1250) — matrix operand order repaired and checked by reading, not compiled, not run on hardware
//
#include <hip/hip_runtime.h>
#include <hip/hip_bf16.h>

// ---------------------------------------------------------------------------
// SGN forward for MI455X (gfx1250, wave32, WMMA).
// All heavy 1x1 convs run as f16 WMMA GEMMs (v_wmma_f32_16x16x32_f16) with
// f32 accumulate and fused bias+BN+ReLU epilogues. Activation layout is
// position-major [pos][C] (pos = (b*V+v)*T+t) so WMMA A/B tiles are contiguous
// 16B/32B per-lane loads matching the CDNA5 16-bit A/B VGPR layouts.
// Each wave register-blocks the full Co strip: A fragment loaded once per
// k-step, reused for Co/16 WMMAs (weights are L0-resident).
// ---------------------------------------------------------------------------

typedef __attribute__((ext_vector_type(16))) _Float16 v16h;
typedef __attribute__((ext_vector_type(8)))  _Float16 v8h;
typedef __attribute__((ext_vector_type(8)))  float    v8f;

#define EPSBN 1e-5f

// ---------------- WMMA tile loaders (CDNA5 16-bit layouts) -----------------
union U16h { v16h v; v8h h[2]; };

// A tile: 16 rows (positions) x 32 K. lanes 0-15: K 0-7 & 16-23; lanes 16-31: K 8-15 & 24-31.
__device__ inline v16h load_a(const _Float16* __restrict__ base, int ldk, int k0, int lane) {
  int row = lane & 15;
  int hi  = lane >> 4;
  const _Float16* p = base + (size_t)row * ldk + k0 + hi * 8;
  U16h u;
  u.h[0] = *(const v8h*)p;
  u.h[1] = *(const v8h*)(p + 16);
  return u.v;
}

// B tile: 32 K x 16 cols (cout). lane = N; lanes 0-15: K 0-7 & 16-23; lanes 16-31: K 8-15 & 24-31.
__device__ inline v16h load_b(const _Float16* __restrict__ wbase, int ldk, int k0, int lane) {
  int n  = lane & 15;
  int hi = lane >> 4;
  const _Float16* p = wbase + (size_t)n * ldk + k0 + hi * 8;
  U16h u;
  u.h[0] = *(const v8h*)p;
  u.h[1] = *(const v8h*)(p + 16);
  return u.v;
}

// ---------------- fused dual-input WMMA GEMM ------------------------------
// out[pos,co] = relu( BN( sum_ci A1[pos,ci]W1[co,ci] (+ sum A2 W2) + bias[co] ) )
// One wave computes a 16 x (COT*16) strip; grid = P/16, block = 32 (EXEC all 1s).
template <int COT>
__global__ void k_wmma_gemm(const _Float16* __restrict__ A1, const _Float16* __restrict__ W1, int K1,
                            const _Float16* __restrict__ A2, const _Float16* __restrict__ W2, int K2,
                            const float* __restrict__ bias, const float* __restrict__ bn,
                            _Float16* __restrict__ out16)
{
  const int  Co      = COT * 16;
  const int  lane    = threadIdx.x;
  const long posBase = (long)blockIdx.x * 16;

  v8f acc[COT];
#pragma unroll
  for (int ct = 0; ct < COT; ++ct) acc[ct] = (v8f){};

  {
    const _Float16* a0 = A1 + posBase * K1;
    for (int k0 = 0; k0 < K1; k0 += 32) {
      v16h av = load_a(a0, K1, k0, lane);
#pragma unroll
      for (int ct = 0; ct < COT; ++ct) {
        v16h bv = load_b(W1 + (size_t)(ct * 16) * K1, K1, k0, lane);
        acc[ct] = __builtin_amdgcn_wmma_f32_16x16x32_f16(false, av, false, bv,
                                                         (short)0, acc[ct], false, false);
      }
    }
  }
  if (K2 > 0) {
    const _Float16* a0 = A2 + posBase * K2;
    for (int k0 = 0; k0 < K2; k0 += 32) {
      v16h av = load_a(a0, K2, k0, lane);
#pragma unroll
      for (int ct = 0; ct < COT; ++ct) {
        v16h bv = load_b(W2 + (size_t)(ct * 16) * K2, K2, k0, lane);
        acc[ct] = __builtin_amdgcn_wmma_f32_16x16x32_f16(false, av, false, bv,
                                                         (short)0, acc[ct], false, false);
      }
    }
  }

  const int n  = lane & 15;
  const int hi = lane >> 4;
#pragma unroll
  for (int ct = 0; ct < COT; ++ct) {
    const int co = ct * 16 + n;
    float bsv = bias ? bias[co] : 0.0f;
    float sc = 1.0f, sh = 0.0f;
    if (bn) {
      float g = bn[co], be = bn[Co + co], mu = bn[2 * Co + co], va = bn[3 * Co + co];
      sc = g * rsqrtf(va + EPSBN);
      sh = be - mu * sc;
    }
#pragma unroll
    for (int r = 0; r < 8; ++r) {
      long pos = posBase + r + hi * 8;   // C/D layout: VGPR r -> M=r (lo lanes) / r+8 (hi lanes)
      float v = (acc[ct][r] + bsv) * sc + sh;
      v = v > 0.0f ? v : 0.0f;
      out16[pos * Co + co] = (_Float16)v;
    }
  }
}

// ---------------- weight conversion ---------------------------------------
__global__ void k_cvt(const float* __restrict__ in, _Float16* __restrict__ out, int n) {
  int i = blockIdx.x * 256 + threadIdx.x;
  if (i < n) out[i] = (_Float16)in[i];
}

// loc_w1 (128,128,3) -> f16 [co][d*128+ci] (tap-major K, matches im2col)
__global__ void k_cvt_locw1(const float* __restrict__ in, _Float16* __restrict__ out) {
  int i = blockIdx.x * 256 + threadIdx.x;
  if (i < 128 * 384) {
    int co = i / 384, r = i % 384, d = r / 128, ci = r % 128;
    out[i] = (_Float16)in[(co * 128 + ci) * 3 + d];
  }
}

// ---------------- one-hot embedding tables (spa1 / tem1) -------------------
// h1 = relu(w1[:,x] + b1); out[x] = relu(w2 @ h1 + b2)
__global__ void k_onehot_embed(const float* __restrict__ w1, const float* __restrict__ b1,
                               const float* __restrict__ w2, const float* __restrict__ b2,
                               int Kin, int Cout, float* __restrict__ out)
{
  __shared__ float h1[64];
  int x = blockIdx.x, tid = threadIdx.x;
  if (tid < 64) {
    float v = w1[tid * Kin + x] + b1[tid];
    h1[tid] = v > 0.0f ? v : 0.0f;
  }
  __syncthreads();
  for (int o = tid; o < Cout; o += blockDim.x) {
    float a = b2[o];
    for (int j = 0; j < 64; ++j) a += w2[o * 64 + j] * h1[j];
    out[x * Cout + o] = a > 0.0f ? a : 0.0f;
  }
}

// ---------------- input embed: transpose + bn + dif + 2x MLP + spa concat --
__global__ void k_input_embed(const float* __restrict__ x,
                              const float* __restrict__ bnj, const float* __restrict__ bnd,
                              const float* __restrict__ jw1, const float* __restrict__ jb1,
                              const float* __restrict__ jw2, const float* __restrict__ jb2,
                              const float* __restrict__ dw1, const float* __restrict__ db1,
                              const float* __restrict__ dw2, const float* __restrict__ db2,
                              const float* __restrict__ spa_tab,
                              _Float16* __restrict__ h0)
{
  __shared__ float s_jw1[192], s_jb1[64], s_jw2[2048], s_jb2[32];
  __shared__ float s_dw1[192], s_db1[64], s_dw2[2048], s_db2[32];
  int tid = threadIdx.x;
  for (int i = tid; i < 192;  i += 256) { s_jw1[i] = jw1[i]; s_dw1[i] = dw1[i]; }
  for (int i = tid; i < 64;   i += 256) { s_jb1[i] = jb1[i]; s_db1[i] = db1[i]; }
  for (int i = tid; i < 2048; i += 256) { s_jw2[i] = jw2[i]; s_dw2[i] = dw2[i]; }
  for (int i = tid; i < 32;   i += 256) { s_jb2[i] = jb2[i]; s_db2[i] = db2[i]; }
  __syncthreads();

  long gid = (long)blockIdx.x * 256 + tid;            // (b,v,t), P == grid*256 exactly
  int t = gid % 64;
  long bv = gid / 64;
  int v = bv % 25;
  int b = bv / 25;
  int nb = b >> 1, m = b & 1;

  float jn[3], dn[3];
#pragma unroll
  for (int c = 0; c < 3; ++c) {
    long xi = ((((long)nb * 3 + c) * 64 + t) * 25 + v) * 2 + m;
    float xc = x[xi];
    float xp = (t > 0) ? x[((((long)nb * 3 + c) * 64 + (t - 1)) * 25 + v) * 2 + m] : xc;
    int ch = c * 25 + v;
    float g = bnj[ch], be = bnj[75 + ch], mu = bnj[150 + ch], va = bnj[225 + ch];
    float s = g * rsqrtf(va + EPSBN);
    jn[c] = (xc - mu) * s + be;
    g = bnd[ch]; be = bnd[75 + ch]; mu = bnd[150 + ch]; va = bnd[225 + ch];
    s = g * rsqrtf(va + EPSBN);
    dn[c] = ((xc - xp) - mu) * s + be;
  }

  float hj[64], hd[64];
#pragma unroll
  for (int j = 0; j < 64; ++j) {
    float a = s_jw1[j * 3] * jn[0] + s_jw1[j * 3 + 1] * jn[1] + s_jw1[j * 3 + 2] * jn[2] + s_jb1[j];
    hj[j] = a > 0.0f ? a : 0.0f;
    float d = s_dw1[j * 3] * dn[0] + s_dw1[j * 3 + 1] * dn[1] + s_dw1[j * 3 + 2] * dn[2] + s_db1[j];
    hd[j] = d > 0.0f ? d : 0.0f;
  }

  _Float16* dst = h0 + gid * 64;
#pragma unroll
  for (int o = 0; o < 32; ++o) {
    float a = s_jb2[o], d = s_db2[o];
    for (int j = 0; j < 64; ++j) { a += s_jw2[o * 64 + j] * hj[j]; d += s_dw2[o * 64 + j] * hd[j]; }
    a = a > 0.0f ? a : 0.0f;
    d = d > 0.0f ? d : 0.0f;
    dst[o] = (_Float16)(a + d);
  }
#pragma unroll
  for (int o = 0; o < 32; ++o) dst[32 + o] = (_Float16)spa_tab[v * 32 + o];
}

// ---------------- attention: q/k projections + 25x25 softmax per (b,t) -----
__global__ void k_attn(const _Float16* __restrict__ h0,
                       const float* __restrict__ g1w, const float* __restrict__ g1b,
                       const float* __restrict__ g2w, const float* __restrict__ g2b,
                       float* __restrict__ gout)
{
  __shared__ _Float16 sh[25 * 64];
  __shared__ float sq[25 * 128], sk[25 * 128];
  __shared__ float sS[25 * 26];
  int tid = threadIdx.x;
  long bt = blockIdx.x;
  int t = bt % 64;
  long b = bt / 64;

  for (int i = tid; i < 1600; i += 256)
    sh[i] = h0[(((b * 25 + (i >> 6)) * 64) + t) * 64 + (i & 63)];
  __syncthreads();

  for (int e = tid; e < 3200; e += 256) {
    int v = e >> 7, o = e & 127;
    float a1 = g1b[o], a2 = g2b[o];
    const _Float16* hv = &sh[v * 64];
    const float* w1r = &g1w[o * 64];
    const float* w2r = &g2w[o * 64];
    for (int c = 0; c < 64; ++c) {
      float hh = (float)hv[c];
      a1 += w1r[c] * hh;
      a2 += w2r[c] * hh;
    }
    sq[v * 128 + o] = a1;
    sk[v * 128 + o] = a2;
  }
  __syncthreads();

  for (int e = tid; e < 625; e += 256) {
    int v = e / 25, u = e % 25;
    float a = 0.0f;
    for (int c = 0; c < 128; ++c) a += sq[v * 128 + c] * sk[u * 128 + c];
    sS[v * 26 + u] = a;
  }
  __syncthreads();

  if (tid < 25) {
    int v = tid;
    float mx = -1e30f;
    for (int u = 0; u < 25; ++u) mx = fmaxf(mx, sS[v * 26 + u]);
    float sum = 0.0f;
    for (int u = 0; u < 25; ++u) { float ex = __expf(sS[v * 26 + u] - mx); sS[v * 26 + u] = ex; sum += ex; }
    float inv = 1.0f / sum;
    for (int u = 0; u < 25; ++u) sS[v * 26 + u] *= inv;
  }
  __syncthreads();

  for (int e = tid; e < 625; e += 256)
    gout[bt * 625 + e] = sS[(e / 25) * 26 + (e % 25)];
}

// ---------------- GCN aggregation: xp[v,c] = sum_u g[v,u] x[u,c] per (b,t) --
__global__ void k_agg(const float* __restrict__ g, const _Float16* __restrict__ xin,
                      _Float16* __restrict__ xp, int C)
{
  __shared__ float sG[625];
  __shared__ _Float16 sX[25 * 128];
  int tid = threadIdx.x;
  long bt = blockIdx.x;
  int t = bt % 64;
  long b = bt / 64;
  for (int i = tid; i < 625; i += 256) sG[i] = g[bt * 625 + i];
  for (int i = tid; i < 25 * C; i += 256)
    sX[i] = xin[((b * 25 + i / C) * 64 + t) * C + (i % C)];
  __syncthreads();
  for (int e = tid; e < 25 * C; e += 256) {
    int v = e / C, c = e % C;
    float a = 0.0f;
    for (int u = 0; u < 25; ++u) a += sG[v * 25 + u] * (float)sX[u * C + c];
    xp[((b * 25 + v) * 64 + t) * C + c] = (_Float16)a;
  }
}

// ---------------- h + tem1, max-pool T windows of 4 ------------------------
__global__ void k_pool(const _Float16* __restrict__ g3, const float* __restrict__ temtab,
                       _Float16* __restrict__ hp)
{
  long i = (long)blockIdx.x * 256 + threadIdx.x;   // (bv, s, c) -> 13,107,200
  int c = i & 127;
  long r = i >> 7;
  int s = r & 15;
  long bv = r >> 4;
  float mx = -1e30f;
#pragma unroll
  for (int d = 0; d < 4; ++d) {
    int t = s * 4 + d;
    float v = (float)g3[(bv * 64 + t) * 128 + c] + temtab[t * 128 + c];
    mx = fmaxf(mx, v);
  }
  hp[i] = (_Float16)mx;
}

// ---------------- im2col for the 1x3 local conv ----------------------------
__global__ void k_im2col(const _Float16* __restrict__ hp, _Float16* __restrict__ col)
{
  long i = (long)blockIdx.x * 256 + threadIdx.x;   // PP2*384
  int k = i % 384;
  long p = i / 384;
  int s = p & 15;
  long bv = p >> 4;
  int d = k / 128, ci = k % 128;
  int ss = s + d - 1;
  _Float16 v = (_Float16)0.0f;
  if (ss >= 0 && ss < 16) v = hp[(bv * 16 + ss) * 128 + ci];
  col[i] = v;
}

// ---------------- global + body-part pooling -------------------------------
__global__ void k_final_pool(const _Float16* __restrict__ f, float* __restrict__ pooled,
                             float* __restrict__ pm)
{
  const unsigned HEADM = (1u << 2) | (1u << 3) | (1u << 20);
  const unsigned HANDM = (1u << 4) | (1u << 5) | (1u << 6) | (1u << 7) | (1u << 8) | (1u << 9) |
                         (1u << 10) | (1u << 11) | (1u << 21) | (1u << 22) | (1u << 23) | (1u << 24);
  const unsigned FOOTM = (1u << 12) | (1u << 13) | (1u << 14) | (1u << 15) | (1u << 16) |
                         (1u << 17) | (1u << 18) | (1u << 19);
  const unsigned HIPM  = (1u << 0) | (1u << 1) | (1u << 2) | (1u << 12) | (1u << 16);
  int n = blockIdx.x, c = threadIdx.x;
  float tot = 0.0f, ph = 0.0f, pa = 0.0f, pf = 0.0f, pp = 0.0f;
  for (int m = 0; m < 2; ++m) {
    long b = (long)n * 2 + m;
    for (int v = 0; v < 25; ++v) {
      float sv = 0.0f;
      for (int s = 0; s < 16; ++s) sv += (float)f[((b * 25 + v) * 16 + s) * 256 + c];
      tot += sv;
      if (HEADM & (1u << v)) ph += sv;
      if (HANDM & (1u << v)) pa += sv;
      if (FOOTM & (1u << v)) pf += sv;
      if (HIPM  & (1u << v)) pp += sv;
    }
  }
  pooled[n * 256 + c] = tot * (1.0f / 800.0f);
  pm[0 * 128 * 256 + n * 256 + c] = ph * (1.0f / (2.0f * 3.0f * 16.0f));
  pm[1 * 128 * 256 + n * 256 + c] = pa * (1.0f / (2.0f * 12.0f * 16.0f));
  pm[2 * 128 * 256 + n * 256 + c] = pf * (1.0f / (2.0f * 8.0f * 16.0f));
  pm[3 * 128 * 256 + n * 256 + c] = pp * (1.0f / (2.0f * 5.0f * 16.0f));
}

// ---------------- classification / projection heads ------------------------
__global__ void k_heads(const float* __restrict__ pooled, const float* __restrict__ pm,
                        const float* __restrict__ fc_w, const float* __restrict__ fc_b,
                        const float* __restrict__ head_w, const float* __restrict__ head_b,
                        const float* __restrict__ part_w, const float* __restrict__ part_b,
                        const float* __restrict__ lscale, float* __restrict__ out)
{
  __shared__ float sv[256];
  int n = blockIdx.x, y = blockIdx.y, tid = threadIdx.x;
  const float *src, *W, *bb;
  int Oc;
  long ofs;
  if (y == 0)      { src = pooled + n * 256; W = fc_w;   bb = fc_b;   Oc = 60;  ofs = 0; }
  else if (y == 1) { src = pooled + n * 256; W = head_w; bb = head_b; Oc = 512; ofs = 7680; }
  else {
    int pi   = (y == 2) ? 0 : (y == 3) ? 1 : (y == 4) ? 3 : 2;          // head, hand, hip, foot
    long oofs = (y == 2) ? 73221 : (y == 3) ? 138757 : (y == 4) ? 204293 : 269829;
    src = pm + (long)pi * 128 * 256 + n * 256;
    W   = part_w + (long)pi * 512 * 256;
    bb  = part_b + pi * 512;
    Oc = 512; ofs = oofs;
  }
  sv[tid] = src[tid];
  __syncthreads();
  for (int o = tid; o < Oc; o += 256) {
    float a = bb[o];
    const float* wr = W + (long)o * 256;
    for (int c = 0; c < 256; ++c) a += sv[c] * wr[c];
    out[ofs + (long)n * Oc + o] = a;
  }
  if (y == 0 && n == 0 && tid < 5) out[73216 + tid] = lscale[tid];
}

// ---------------------------------------------------------------------------
extern "C" void kernel_launch(void* const* d_in, const int* in_sizes, int n_in,
                              void* d_out, int out_size, void* d_ws, size_t ws_size,
                              hipStream_t stream)
{
  const float* x        = (const float*)d_in[0];
  const float* bn_joint = (const float*)d_in[1];
  const float* joint_w1 = (const float*)d_in[2];
  const float* joint_b1 = (const float*)d_in[3];
  const float* joint_w2 = (const float*)d_in[4];
  const float* joint_b2 = (const float*)d_in[5];
  const float* bn_dif   = (const float*)d_in[6];
  const float* dif_w1   = (const float*)d_in[7];
  const float* dif_b1   = (const float*)d_in[8];
  const float* dif_w2   = (const float*)d_in[9];
  const float* dif_b2   = (const float*)d_in[10];
  const float* tem_w1   = (const float*)d_in[11];
  const float* tem_b1   = (const float*)d_in[12];
  const float* tem_w2   = (const float*)d_in[13];
  const float* tem_b2   = (const float*)d_in[14];
  const float* spa_w1   = (const float*)d_in[15];
  const float* spa_b1   = (const float*)d_in[16];
  const float* spa_w2   = (const float*)d_in[17];
  const float* spa_b2   = (const float*)d_in[18];
  const float* g1_w     = (const float*)d_in[19];
  const float* g1_b     = (const float*)d_in[20];
  const float* g2_w     = (const float*)d_in[21];
  const float* g2_b     = (const float*)d_in[22];
  const float* gcn1_w   = (const float*)d_in[23];
  const float* gcn1_w1  = (const float*)d_in[24];
  const float* gcn1_b1  = (const float*)d_in[25];
  const float* bn_gcn1  = (const float*)d_in[26];
  const float* gcn2_w   = (const float*)d_in[27];
  const float* gcn2_w1  = (const float*)d_in[28];
  const float* gcn2_b1  = (const float*)d_in[29];
  const float* bn_gcn2  = (const float*)d_in[30];
  const float* gcn3_w   = (const float*)d_in[31];
  const float* gcn3_w1  = (const float*)d_in[32];
  const float* gcn3_b1  = (const float*)d_in[33];
  const float* bn_gcn3  = (const float*)d_in[34];
  const float* loc_w1   = (const float*)d_in[35];
  const float* loc_b1   = (const float*)d_in[36];
  const float* bn_loc1  = (const float*)d_in[37];
  const float* loc_w2   = (const float*)d_in[38];
  const float* loc_b2   = (const float*)d_in[39];
  const float* bn_loc2  = (const float*)d_in[40];
  const float* fc_w     = (const float*)d_in[41];
  const float* fc_b     = (const float*)d_in[42];
  const float* head_w   = (const float*)d_in[43];
  const float* head_b   = (const float*)d_in[44];
  const float* part_w   = (const float*)d_in[45];
  const float* part_b   = (const float*)d_in[46];
  const float* lscale   = (const float*)d_in[47];

  char* ws = (char*)d_ws;
  // big activation slots (lifetime-based reuse)
  size_t o_h0  = 0;                          // h0 f16 (P*64)        -> later hp f16 (P2*128)
  size_t o_g   = o_h0  + 52428800;           // attention g f32 (B*T*625)
  size_t o_xp  = o_g   + 40960000;           // xp f16 (P*128 max)   -> later im2col (P2*384)
  size_t o_g1o = o_xp  + 104857600;          // gcn1 out f16 (P*64)  -> later loc1 out (P2*128)
  size_t o_g2o = o_g1o + 52428800;           // gcn2 out f16 (P*128) -> later loc2 out (P2*256)
  size_t o_g3o = o_g2o + 104857600;          // gcn3 out f16 (P*128)
  size_t o_w16 = o_g3o + 104857600;          // f16 weights + tables + pooled vecs

  _Float16* h0    = (_Float16*)(ws + o_h0);
  float*    gbuf  = (float*)   (ws + o_g);
  _Float16* xp    = (_Float16*)(ws + o_xp);
  _Float16* g1o   = (_Float16*)(ws + o_g1o);
  _Float16* g2o   = (_Float16*)(ws + o_g2o);
  _Float16* g3o   = (_Float16*)(ws + o_g3o);
  _Float16* hp    = h0;                       // reuse: h0 dead after gcn1
  _Float16* col   = xp;                       // reuse: xp dead after gcn3
  _Float16* loc1o = g1o;                      // reuse: gcn1o dead after gcn2
  _Float16* loc2o = g2o;                      // reuse: gcn2o dead after gcn3

  size_t cur = o_w16;
  auto halfSlot = [&](size_t nh) { _Float16* p = (_Float16*)(ws + cur); cur += ((nh * 2 + 255) & ~255ULL); return p; };
  auto f32Slot  = [&](size_t nf) { float* p = (float*)(ws + cur); cur += ((nf * 4 + 255) & ~255ULL); return p; };
  _Float16* w_g1   = halfSlot(64 * 64);
  _Float16* w_g1b  = halfSlot(64 * 64);       // gcn1_w1
  _Float16* w_g2   = halfSlot(128 * 64);
  _Float16* w_g2b  = halfSlot(128 * 64);
  _Float16* w_g3   = halfSlot(128 * 128);
  _Float16* w_g3b  = halfSlot(128 * 128);
  _Float16* w_l1   = halfSlot(128 * 384);
  _Float16* w_l2   = halfSlot(256 * 128);
  float*    spaTab = f32Slot(25 * 32);
  float*    temTab = f32Slot(64 * 128);
  float*    pooled = f32Slot(128 * 256);
  float*    pm     = f32Slot(4 * 128 * 256);

  // 1) weight conversions + one-hot tables
  k_cvt<<<dim3(16),  dim3(256), 0, stream>>>(gcn1_w,  w_g1,  64 * 64);
  k_cvt<<<dim3(16),  dim3(256), 0, stream>>>(gcn1_w1, w_g1b, 64 * 64);
  k_cvt<<<dim3(32),  dim3(256), 0, stream>>>(gcn2_w,  w_g2,  128 * 64);
  k_cvt<<<dim3(32),  dim3(256), 0, stream>>>(gcn2_w1, w_g2b, 128 * 64);
  k_cvt<<<dim3(64),  dim3(256), 0, stream>>>(gcn3_w,  w_g3,  128 * 128);
  k_cvt<<<dim3(64),  dim3(256), 0, stream>>>(gcn3_w1, w_g3b, 128 * 128);
  k_cvt<<<dim3(128), dim3(256), 0, stream>>>(loc_w2,  w_l2,  256 * 128);
  k_cvt_locw1<<<dim3(192), dim3(256), 0, stream>>>(loc_w1, w_l1);
  k_onehot_embed<<<dim3(25), dim3(128), 0, stream>>>(spa_w1, spa_b1, spa_w2, spa_b2, 25, 32,  spaTab);
  k_onehot_embed<<<dim3(64), dim3(128), 0, stream>>>(tem_w1, tem_b1, tem_w2, tem_b2, 64, 128, temTab);

  // 2) input embed -> h0 (P x 64 f16)
  k_input_embed<<<dim3(1600), dim3(256), 0, stream>>>(
      x, bn_joint, bn_dif, joint_w1, joint_b1, joint_w2, joint_b2,
      dif_w1, dif_b1, dif_w2, dif_b2, spaTab, h0);

  // 3) attention g (B*T x 25 x 25)
  k_attn<<<dim3(16384), dim3(256), 0, stream>>>(h0, g1_w, g1_b, g2_w, g2_b, gbuf);

  // 4) GCN stack (agg -> fused dual-GEMM + BN + ReLU), register-blocked Co strips
  k_agg<<<dim3(16384), dim3(256), 0, stream>>>(gbuf, h0, xp, 64);
  k_wmma_gemm<4><<<dim3(25600), dim3(32), 0, stream>>>(xp, w_g1, 64, h0, w_g1b, 64,
                                                       gcn1_b1, bn_gcn1, g1o);
  k_agg<<<dim3(16384), dim3(256), 0, stream>>>(gbuf, g1o, xp, 64);
  k_wmma_gemm<8><<<dim3(25600), dim3(32), 0, stream>>>(xp, w_g2, 64, g1o, w_g2b, 64,
                                                       gcn2_b1, bn_gcn2, g2o);
  k_agg<<<dim3(16384), dim3(256), 0, stream>>>(gbuf, g2o, xp, 128);
  k_wmma_gemm<8><<<dim3(25600), dim3(32), 0, stream>>>(xp, w_g3, 128, g2o, w_g3b, 128,
                                                       gcn3_b1, bn_gcn3, g3o);

  // 5) + tem1, max-pool T/4 -> hp (P2 x 128)
  k_pool<<<dim3(51200), dim3(256), 0, stream>>>(g3o, temTab, hp);

  // 6) local 1x3 conv via im2col + WMMA, then 1x1 conv to 256
  k_im2col<<<dim3(153600), dim3(256), 0, stream>>>(hp, col);
  k_wmma_gemm<8><<<dim3(6400), dim3(32), 0, stream>>>(col, w_l1, 384, nullptr, nullptr, 0,
                                                      loc_b1, bn_loc1, loc1o);
  k_wmma_gemm<16><<<dim3(6400), dim3(32), 0, stream>>>(loc1o, w_l2, 128, nullptr, nullptr, 0,
                                                       loc_b2, bn_loc2, loc2o);

  // 7) pooling + heads -> d_out
  k_final_pool<<<dim3(128), dim3(256), 0, stream>>>(loc2o, pooled, pm);
  k_heads<<<dim3(128, 6), dim3(256), 0, stream>>>(pooled, pm, fc_w, fc_b, head_w, head_b,
                                                  part_w, part_b, lscale, (float*)d_out);
}
